// KnnModule_73461120631584
// MI455X (gfx1250) — matrix operand order repaired and checked by reading, not compiled, not run on hardware
//
#include <hip/hip_runtime.h>
#include <math.h>

// ---- problem constants (match reference) ----
#define NQ      2048
#define DIM     1024
#define NTRAIN  100000
#define NCHUNK  10
#define CHUNK   10000          // NCHUNK * CHUNK == NTRAIN
#define CHUNKP  10016          // chunk padded to a multiple of 32 (ghost cols)
#define MAXK    200
#define NCLS    1000
#define INVT    (1.0f / 0.07f)

typedef __attribute__((ext_vector_type(16))) __bf16          v16bf;
typedef __attribute__((ext_vector_type(8)))  float           v8f;

union V16 { v16bf v; uint4 u[2]; };

// ============================================================================
// Kernel 0: fp32 -> bf16 hi (RNE) + bf16 lo (residual, truncated).
// Each thread converts 8 consecutive floats; writes 16B hi + 16B lo.
// ============================================================================
__global__ __launch_bounds__(256) void knn_split_bf16(
    const float* __restrict__ src, unsigned short* __restrict__ hi,
    unsigned short* __restrict__ lo, int n)
{
    int base = (blockIdx.x * 256 + threadIdx.x) * 8;
    if (base >= n) return;
    const float4* p = reinterpret_cast<const float4*>(src + base);
    float4 f0 = p[0], f1 = p[1];
    float f[8] = { f0.x, f0.y, f0.z, f0.w, f1.x, f1.y, f1.z, f1.w };
    unsigned short h[8], l[8];
#pragma unroll
    for (int e = 0; e < 8; ++e) {
        unsigned int b = __float_as_uint(f[e]);
        unsigned int rh = (b + 0x7FFFu + ((b >> 16) & 1u)) >> 16;  // RNE
        h[e] = (unsigned short)rh;
        float hf = __uint_as_float(rh << 16);
        float r  = f[e] - hf;
        l[e] = (unsigned short)(__float_as_uint(r) >> 16);
    }
    union { unsigned short s[8]; uint4 u; } oh, ol;
#pragma unroll
    for (int e = 0; e < 8; ++e) { oh.s[e] = h[e]; ol.s[e] = l[e]; }
    *reinterpret_cast<uint4*>(hi + base) = oh.u;
    *reinterpret_cast<uint4*>(lo + base) = ol.u;
}

// ============================================================================
// Kernel 1: GEMM for one chunk, bf16 hi/lo inputs, 2x2 register blocking.
// Block = 8 waves; each wave owns a 32(row)x32(col) macro-tile = 4 16x16
// tiles. Per 32-wide K step: 16x global_load_b128, 12x wmma_f32_16x16x32_bf16
// (hi fragments feed 4 WMMAs, lo fragments 2). All branches wave-uniform -> EXEC full.
// Fragment addressing for A(16x32)/B(32x16), same k order in both (not ISA table order):
//   A: lane    -> row=lane&15, K = kb + 16*(lane>>4) .. +15   (contiguous)
//   B: lane    -> col=lane&15, K = kb + 16*(lane>>4) .. +15   (contiguous)
// ============================================================================
__global__ __launch_bounds__(256) void knn_gemm_chunk(
    const unsigned short* __restrict__ qh, const unsigned short* __restrict__ ql,
    const unsigned short* __restrict__ th, const unsigned short* __restrict__ tl,
    float* __restrict__ sims)             // [NQ][CHUNKP]
{
    const int wave = threadIdx.x >> 5;
    const int lane = threadIdx.x & 31;
    const int half = lane >> 4;
    const int l16  = lane & 15;

    const int strip = blockIdx.y * 8 + wave;       // 32-col strip (0..312)
    if (strip * 32 >= CHUNKP) return;              // wave-uniform guard

    const int rowBase  = blockIdx.x * 32;
    const int colLocal = strip * 32;

    const unsigned short* aH0 = qh + (size_t)(rowBase + l16) * DIM + half * 16;
    const unsigned short* aL0 = ql + (size_t)(rowBase + l16) * DIM + half * 16;
    const unsigned short* aH1 = aH0 + (size_t)16 * DIM;
    const unsigned short* aL1 = aL0 + (size_t)16 * DIM;
    const unsigned short* bH0 = th + (size_t)(colLocal + l16) * DIM + half * 16;
    const unsigned short* bL0 = tl + (size_t)(colLocal + l16) * DIM + half * 16;
    const unsigned short* bH1 = bH0 + (size_t)16 * DIM;
    const unsigned short* bL1 = bL0 + (size_t)16 * DIM;

    v8f c00 = {}, c01 = {}, c10 = {}, c11 = {};
    for (int kb = 0; kb < DIM; kb += 32) {
        V16 Ah0, Al0, Ah1, Al1, Bh0, Bl0, Bh1, Bl1;
        Ah0.u[0] = *reinterpret_cast<const uint4*>(aH0 + kb);
        Ah0.u[1] = *reinterpret_cast<const uint4*>(aH0 + kb + 8);
        Al0.u[0] = *reinterpret_cast<const uint4*>(aL0 + kb);
        Al0.u[1] = *reinterpret_cast<const uint4*>(aL0 + kb + 8);
        Ah1.u[0] = *reinterpret_cast<const uint4*>(aH1 + kb);
        Ah1.u[1] = *reinterpret_cast<const uint4*>(aH1 + kb + 8);
        Al1.u[0] = *reinterpret_cast<const uint4*>(aL1 + kb);
        Al1.u[1] = *reinterpret_cast<const uint4*>(aL1 + kb + 8);
        Bh0.u[0] = *reinterpret_cast<const uint4*>(bH0 + kb);
        Bh0.u[1] = *reinterpret_cast<const uint4*>(bH0 + kb + 8);
        Bl0.u[0] = *reinterpret_cast<const uint4*>(bL0 + kb);
        Bl0.u[1] = *reinterpret_cast<const uint4*>(bL0 + kb + 8);
        Bh1.u[0] = *reinterpret_cast<const uint4*>(bH1 + kb);
        Bh1.u[1] = *reinterpret_cast<const uint4*>(bH1 + kb + 8);
        Bl1.u[0] = *reinterpret_cast<const uint4*>(bL1 + kb);
        Bl1.u[1] = *reinterpret_cast<const uint4*>(bL1 + kb + 8);

        // sim += Ahi*Bhi + Ahi*Blo + Alo*Bhi  per tile (~fp32-accurate split)
        c00 = __builtin_amdgcn_wmma_f32_16x16x32_bf16(false, Ah0.v, false, Bh0.v, (short)0, c00, false, false);
        c00 = __builtin_amdgcn_wmma_f32_16x16x32_bf16(false, Ah0.v, false, Bl0.v, (short)0, c00, false, false);
        c00 = __builtin_amdgcn_wmma_f32_16x16x32_bf16(false, Al0.v, false, Bh0.v, (short)0, c00, false, false);
        c01 = __builtin_amdgcn_wmma_f32_16x16x32_bf16(false, Ah0.v, false, Bh1.v, (short)0, c01, false, false);
        c01 = __builtin_amdgcn_wmma_f32_16x16x32_bf16(false, Ah0.v, false, Bl1.v, (short)0, c01, false, false);
        c01 = __builtin_amdgcn_wmma_f32_16x16x32_bf16(false, Al0.v, false, Bh1.v, (short)0, c01, false, false);
        c10 = __builtin_amdgcn_wmma_f32_16x16x32_bf16(false, Ah1.v, false, Bh0.v, (short)0, c10, false, false);
        c10 = __builtin_amdgcn_wmma_f32_16x16x32_bf16(false, Ah1.v, false, Bl0.v, (short)0, c10, false, false);
        c10 = __builtin_amdgcn_wmma_f32_16x16x32_bf16(false, Al1.v, false, Bh0.v, (short)0, c10, false, false);
        c11 = __builtin_amdgcn_wmma_f32_16x16x32_bf16(false, Ah1.v, false, Bh1.v, (short)0, c11, false, false);
        c11 = __builtin_amdgcn_wmma_f32_16x16x32_bf16(false, Ah1.v, false, Bl1.v, (short)0, c11, false, false);
        c11 = __builtin_amdgcn_wmma_f32_16x16x32_bf16(false, Al1.v, false, Bh1.v, (short)0, c11, false, false);
    }

    // C/D layout: VGPR r -> row = r + 8*half, col = l16
#pragma unroll
    for (int r = 0; r < 8; ++r) {
        int row0 = rowBase + r + half * 8;
        int col0 = colLocal + l16;
        sims[(size_t)row0 * CHUNKP + col0]          = c00[r];
        sims[(size_t)row0 * CHUNKP + col0 + 16]     = c01[r];
        sims[(size_t)(row0 + 16) * CHUNKP + col0]      = c10[r];
        sims[(size_t)(row0 + 16) * CHUNKP + col0 + 16] = c11[r];
    }
}

// ============================================================================
// Kernel 2: per-row top-200 of one chunk via 2048-bin radix histogram on the
// order-preserving float->uint map, then threshold gather. One block per row.
// Only the first CHUNK (10000) columns are scanned; padded cols are ignored.
// ============================================================================
__global__ __launch_bounds__(256) void knn_chunk_topk(
    const float* __restrict__ sims,   // [NQ][CHUNKP]
    float* __restrict__ cval,         // [NQ][NCHUNK*MAXK]
    int*   __restrict__ cidx,
    int chunkIdx)
{
    __shared__ int hist[2048];
    __shared__ int gsum[256];
    __shared__ int s_tb, s_above, s_need, s_cntHi, s_cntTie;

    const int tid = threadIdx.x;
    const int row = blockIdx.x;
    const float* sr = sims + (size_t)row * CHUNKP;

    for (int i = tid; i < 2048; i += 256) hist[i] = 0;
    __syncthreads();

    for (int i = tid; i < CHUNK; i += 256) {
        unsigned int b = __float_as_uint(sr[i]);
        unsigned int u = b ^ ((b & 0x80000000u) ? 0xFFFFFFFFu : 0x80000000u);
        atomicAdd(&hist[u >> 21], 1);
    }
    __syncthreads();

    {
        int s = 0;
#pragma unroll
        for (int j = 0; j < 8; ++j) s += hist[tid * 8 + j];
        gsum[tid] = s;
    }
    __syncthreads();

    if (tid == 0) {
        int cum = 0, tb = 0, above = 0;
        for (int g = 255; g >= 0; --g) {
            if (cum + gsum[g] >= MAXK) {
                for (int b = g * 8 + 7; b >= g * 8; --b) {
                    if (cum + hist[b] >= MAXK) { tb = b; above = cum; break; }
                    cum += hist[b];
                }
                break;
            }
            cum += gsum[g];
        }
        s_tb = tb; s_above = above; s_need = MAXK - above;
        s_cntHi = 0; s_cntTie = 0;
    }
    __syncthreads();

    const int tb = s_tb, above = s_above, need = s_need;
    float* ov = cval + (size_t)row * (NCHUNK * MAXK) + chunkIdx * MAXK;
    int*   oi = cidx + (size_t)row * (NCHUNK * MAXK) + chunkIdx * MAXK;
    const int chunkBase = chunkIdx * CHUNK;

    for (int i = tid; i < CHUNK; i += 256) {
        float v = sr[i];
        unsigned int b = __float_as_uint(v);
        unsigned int u = b ^ ((b & 0x80000000u) ? 0xFFFFFFFFu : 0x80000000u);
        int bin = (int)(u >> 21);
        int slot = -1;
        if (bin > tb) {
            slot = atomicAdd(&s_cntHi, 1);                 // always < above
        } else if (bin == tb) {
            int t = atomicAdd(&s_cntTie, 1);
            if (t < need) slot = above + t;
        }
        if (slot >= 0) { ov[slot] = v; oi[slot] = chunkBase + i; }
    }
}

// ============================================================================
// Kernel 3: merge 2000 candidates/row -> sorted top-200, softmax, nested-k
// class histograms. One block (256 threads) per row.
// ============================================================================
__global__ __launch_bounds__(256) void knn_final(
    const float* __restrict__ cval,
    const int*   __restrict__ cidx,
    const int*   __restrict__ labels,
    float*       __restrict__ out)    // [4][NQ][NCLS]
{
    __shared__ float sval[2048];
    __shared__ int   sidx[2048];
    __shared__ int   slab[256];
    __shared__ float cls[NCLS];
    __shared__ float red[256];

    const int tid = threadIdx.x;
    const int row = blockIdx.x;
    const int NC  = NCHUNK * MAXK;    // 2000

    for (int i = tid; i < 2048; i += 256) {
        if (i < NC) {
            sval[i] = cval[(size_t)row * NC + i];
            sidx[i] = cidx[(size_t)row * NC + i];
        } else {
            sval[i] = -INFINITY;
            sidx[i] = 0;
        }
    }
    for (int j = tid; j < NCLS; j += 256) cls[j] = 0.0f;
    __syncthreads();

    // bitonic sort, descending by sval (indices ride along)
    for (int sz = 2; sz <= 2048; sz <<= 1) {
        for (int st = sz >> 1; st > 0; st >>= 1) {
            for (int i = tid; i < 2048; i += 256) {
                int ixj = i ^ st;
                if (ixj > i) {
                    bool desc = ((i & sz) == 0);
                    float vi = sval[i], vj = sval[ixj];
                    bool sw = desc ? (vi < vj) : (vi > vj);
                    if (sw) {
                        int ti = sidx[i];
                        sval[i] = vj;  sval[ixj] = vi;
                        sidx[i] = sidx[ixj]; sidx[ixj] = ti;
                    }
                }
            }
            __syncthreads();
        }
    }

    if (tid < MAXK) slab[tid] = labels[sidx[tid]];

    // softmax over top-200 at temperature T
    float m = sval[0];
    float e = 0.0f;
    if (tid < MAXK) e = __expf((sval[tid] - m) * INVT);
    red[tid] = e;
    __syncthreads();
    for (int s = 128; s > 0; s >>= 1) {
        if (tid < s) red[tid] += red[tid + s];
        __syncthreads();
    }
    float w = e / red[0];
    __syncthreads();

    // nested-k histograms: accumulate incrementally, snapshot after each k
    const int ks[4] = {10, 20, 100, 200};
    int kprev = 0;
    for (int kk = 0; kk < 4; ++kk) {
        int k = ks[kk];
        if (tid >= kprev && tid < k) atomicAdd(&cls[slab[tid]], w);
        __syncthreads();
        float* o = out + ((size_t)kk * NQ + row) * NCLS;
        for (int jj = tid; jj < NCLS; jj += 256) o[jj] = cls[jj];
        __syncthreads();
        kprev = k;
    }
}

// ============================================================================
// Host launcher.  Workspace layout (~165 MB):
//   sims : float [NQ*CHUNKP]        =  82,051,072 B  (reused per chunk)
//   cval : float [NQ*NCHUNK*MAXK]   =  16,384,000 B
//   cidx : int   [NQ*NCHUNK*MAXK]   =  16,384,000 B
//   qh,ql: bf16  [NQ*DIM]           =   4,194,304 B each
//   th,tl: bf16  [CHUNKP*DIM]       =  20,512,768 B each (reused per chunk;
//                                      16 ghost rows never influence output)
// ============================================================================
extern "C" void kernel_launch(void* const* d_in, const int* in_sizes, int n_in,
                              void* d_out, int out_size, void* d_ws, size_t ws_size,
                              hipStream_t stream) {
    (void)in_sizes; (void)n_in; (void)out_size; (void)ws_size;

    const float* q      = (const float*)d_in[0];   // [2048*1024] f32
    const float* tr     = (const float*)d_in[1];   // [100000*1024] f32
    const int*   labels = (const int*)d_in[2];     // [100000] i32
    float*       out    = (float*)d_out;           // [4*2048*1000] f32

    char* ws = (char*)d_ws;
    size_t off = 0;
    float* sims = (float*)(ws + off);           off += (size_t)NQ * CHUNKP * 4;
    float* cval = (float*)(ws + off);           off += (size_t)NQ * NCHUNK * MAXK * 4;
    int*   cidx = (int*)(ws + off);             off += (size_t)NQ * NCHUNK * MAXK * 4;
    unsigned short* qh = (unsigned short*)(ws + off); off += (size_t)NQ * DIM * 2;
    unsigned short* ql = (unsigned short*)(ws + off); off += (size_t)NQ * DIM * 2;
    unsigned short* th = (unsigned short*)(ws + off); off += (size_t)CHUNKP * DIM * 2;
    unsigned short* tl = (unsigned short*)(ws + off);

    // split Q into bf16 hi/lo once per call
    {
        int n = NQ * DIM;
        knn_split_bf16<<<dim3(n / (256 * 8)), 256, 0, stream>>>(q, qh, ql, n);
    }

    dim3 gemmGrid(NQ / 32, (CHUNKP / 32 + 7) / 8);  // 64 x 40
    for (int c = 0; c < NCHUNK; ++c) {
        int n = CHUNK * DIM;
        knn_split_bf16<<<dim3(n / (256 * 8)), 256, 0, stream>>>(
            tr + (size_t)c * CHUNK * DIM, th, tl, n);
        knn_gemm_chunk<<<gemmGrid, 256, 0, stream>>>(qh, ql, th, tl, sims);
        knn_chunk_topk<<<dim3(NQ), 256, 0, stream>>>(sims, cval, cidx, c);
    }
    knn_final<<<dim3(NQ), 256, 0, stream>>>(cval, cidx, labels, out);
}
